// ConditionalFlow_7146825580750
// MI455X (gfx1250) — matrix operand order repaired and checked by reading, not compiled, not run on hardware
//
#include <hip/hip_runtime.h>
#include <hip/hip_bf16.h>

// ---------------------------------------------------------------------------
// Problem sizes (compile-time constants from the reference)
// ---------------------------------------------------------------------------
#define BSZ    8192
#define NOISE  512
#define LATENT 2048
#define CONDD  512
#define NBLK   8
#define NCLS   1024

// GEMM tiling: 128x128x32, 256 threads = 8 wave32s (4 along M, 2 along N)
#define BM 128
#define BN 128
#define BK 32
#define LDSTRIDE 40   // padded row stride in bf16 elements (80B, 16B aligned)

typedef __attribute__((ext_vector_type(16))) __bf16          bf16x16;
typedef __attribute__((ext_vector_type(8)))  float           v8f;
typedef __attribute__((ext_vector_type(8)))  unsigned short  v8us;

union FragAB { v8us h[2]; bf16x16 v; };

__device__ __forceinline__ unsigned short f2bf(float f) {
    unsigned int u = __builtin_bit_cast(unsigned int, f);
    unsigned int r = u + 0x7FFFu + ((u >> 16) & 1u);   // round-to-nearest-even
    return (unsigned short)(r >> 16);
}

// SiLU with hardware reciprocal (v_rcp_f32) instead of IEEE divide sequence;
// result feeds bf16 rounding, so the ~1ulp rcp approximation is invisible.
__device__ __forceinline__ float silu(float v) {
    return v * __builtin_amdgcn_rcpf(1.0f + __expf(-v));
}

// ---------------------------------------------------------------------------
// WMMA GEMM:  out = epilogue(A[MxK] @ B[KxN] + bias)
//   MODE 0: out bf16 = silu(acc + bias)                         (cond/mlp fc1)
//   MODE 1: out f32  = acc + bias                               (c tensor)
//   MODE 2: A = silu(gather(emb_table, idx)); out bf16 = acc+bias+t_emb
//   MODE 3: out f32  = res + (acc + bias) * (1 + scale2) / NBLK (mlp fc2 + res)
// A is bf16 row-major (stride K) except MODE 2 (f32 table gather).
// B is f32 row-major [K x N]; converted to bf16 while staging into LDS.
// ---------------------------------------------------------------------------
template <int MODE>
__global__ __launch_bounds__(256) void gemm_wmma(
    const void*  __restrict__ Aptr,
    const float* __restrict__ Bw,
    const float* __restrict__ bias,
    void*        __restrict__ outp,
    int M, int N, int K,
    const int*   __restrict__ idx,    // MODE 2: class indices
    const float* __restrict__ timev,  // MODE 2: time [B]
    const float* __restrict__ cvec,   // MODE 3: c tensor [B,3*NOISE]
    const float* __restrict__ resv)   // MODE 3: residual x [B,NOISE]
{
    __shared__ alignas(16) unsigned short lA[BM * LDSTRIDE];
    __shared__ alignas(16) unsigned short lB[BN * LDSTRIDE];

    const int t    = threadIdx.x;
    const int m0   = blockIdx.y * BM;
    const int n0   = blockIdx.x * BN;
    const int lane = t & 31;
    const int wid  = t >> 5;
    const int waveM = wid & 3;        // 4 waves along M (32 rows each)
    const int waveN = wid >> 2;       // 2 waves along N (64 cols each)
    const int half  = lane >> 4;      // lane group (K-split per ISA layout)
    const int lm    = lane & 15;

    // global-load assignments (each thread moves 16 elements per tile)
    const int aRow = t >> 1, aCol = (t & 1) * 16;   // A tile: 128 x 32
    const int bRow = t >> 3, bCol = (t & 7) * 16;   // B tile: 32 x 128

    const unsigned short* hA = (const unsigned short*)Aptr;
    const int arow_g = m0 + aRow;

    const float* aSrcF = nullptr;
    if constexpr (MODE == 2)
        aSrcF = (const float*)Aptr + (size_t)idx[arow_g] * K;

    v8f zero;
#pragma unroll
    for (int r = 0; r < 8; ++r) zero[r] = 0.0f;
    v8f acc[2][4];
#pragma unroll
    for (int i = 0; i < 2; ++i)
#pragma unroll
        for (int j = 0; j < 4; ++j) acc[i][j] = zero;

    for (int kt = 0; kt < K; kt += BK) {
        // ---- stage A tile (bf16, row-major, K contiguous) ----
        if constexpr (MODE == 2) {
            const float* p = aSrcF + kt + aCol;
#pragma unroll
            for (int j = 0; j < 16; ++j)
                lA[aRow * LDSTRIDE + aCol + j] = f2bf(silu(p[j]));
        } else {
            const v8us* p = (const v8us*)(hA + (size_t)arow_g * K + kt + aCol);
            *(v8us*)&lA[aRow * LDSTRIDE + aCol]     = p[0];
            *(v8us*)&lA[aRow * LDSTRIDE + aCol + 8] = p[1];
        }
        // ---- stage B tile transposed: lB[n][k], f32 -> bf16 ----
        {
            const float* p = Bw + (size_t)(kt + bRow) * N + n0 + bCol;
#pragma unroll
            for (int j = 0; j < 16; ++j)
                lB[(bCol + j) * LDSTRIDE + bRow] = f2bf(p[j]);
        }
        __syncthreads();

        // prefetch next global tiles (global_prefetch_b8 on gfx1250)
        if (kt + BK < K) {
            if constexpr (MODE != 2)
                __builtin_prefetch(hA + (size_t)arow_g * K + kt + BK + aCol, 0, 0);
            else
                __builtin_prefetch(aSrcF + kt + BK + aCol, 0, 0);
            __builtin_prefetch(Bw + (size_t)(kt + BK + bRow) * N + n0 + bCol, 0, 0);
        }

        // ---- build fragments per ISA lane layout ----
        // A 16x32 bf16: lane holds M=lm; K = 8*half + e (e<8), 16+8*half+(e-8)
        FragAB fa[2];
#pragma unroll
        for (int sm = 0; sm < 2; ++sm) {
            const v8us* pa = (const v8us*)&lA[(waveM * 32 + sm * 16 + lm) * LDSTRIDE];
            fa[sm].h[0] = pa[half];
            fa[sm].h[1] = pa[2 + half];
        }
        // B 32x16 bf16: lane holds N=lm; K = 8*half + e (e<8), 16+8*half+(e-8)
        FragAB fb[4];
#pragma unroll
        for (int sn = 0; sn < 4; ++sn) {
            const v8us* pb = (const v8us*)&lB[(waveN * 64 + sn * 16 + lm) * LDSTRIDE];
            fb[sn].h[0] = pb[half];
            fb[sn].h[1] = pb[2 + half];
        }
#pragma unroll
        for (int sm = 0; sm < 2; ++sm)
#pragma unroll
            for (int sn = 0; sn < 4; ++sn)
                acc[sm][sn] = __builtin_amdgcn_wmma_f32_16x16x32_bf16(
                    false, fa[sm].v, false, fb[sn].v,
                    (short)0, acc[sm][sn], false, false);
        __syncthreads();
    }

    // ---- epilogue; C layout: VGPR r -> M = r + 8*half, N = lm ----
#pragma unroll
    for (int sm = 0; sm < 2; ++sm) {
#pragma unroll
        for (int sn = 0; sn < 4; ++sn) {
#pragma unroll
            for (int r = 0; r < 8; ++r) {
                const int m = m0 + waveM * 32 + sm * 16 + half * 8 + r;
                const int n = n0 + waveN * 64 + sn * 16 + lm;
                float v = acc[sm][sn][r] + bias[n];
                const size_t o = (size_t)m * N + n;
                if constexpr (MODE == 0) {
                    ((unsigned short*)outp)[o] = f2bf(silu(v));
                } else if constexpr (MODE == 1) {
                    ((float*)outp)[o] = v;
                } else if constexpr (MODE == 2) {
                    // sinusoidal embedding: freqs = 10^linspace(0,3,256)
                    const float tt = timev[m];
                    const float j  = (float)((n < 256) ? n : (n - 256));
                    const float fr = exp2f(j * (3.0f / 255.0f) * 3.3219280948873623f);
                    const float ang = 6.283185307179586f * fr * tt;
                    v += (n < 256) ? __sinf(ang) : __cosf(ang);
                    ((unsigned short*)outp)[o] = f2bf(v);
                } else {
                    const float s2 = cvec[(size_t)m * (3 * NOISE) + 2 * NOISE + n];
                    ((float*)outp)[o] = resv[o] + v * (1.0f + s2) * (1.0f / NBLK);
                }
            }
        }
    }
}

// ---------------------------------------------------------------------------
// LayerNorm + AdaLN modulate:  xm = (1+scale1) * LN(x) + shift   (bf16 out)
// One 256-thread block per row of 512.
// ---------------------------------------------------------------------------
__global__ __launch_bounds__(256) void ln_mod_kernel(
    const float* __restrict__ x, const float* __restrict__ c,
    unsigned short* __restrict__ xm)
{
    const int row = blockIdx.x;
    const int t   = threadIdx.x;
    const float* xr = x + (size_t)row * NOISE;
    const float v0 = xr[t], v1 = xr[t + 256];

    __shared__ float ss[256], sq[256];
    ss[t] = v0 + v1;
    sq[t] = v0 * v0 + v1 * v1;
    __syncthreads();
#pragma unroll
    for (int o = 128; o > 0; o >>= 1) {
        if (t < o) { ss[t] += ss[t + o]; sq[t] += sq[t + o]; }
        __syncthreads();
    }
    const float mu  = ss[0] * (1.0f / NOISE);
    const float var = sq[0] * (1.0f / NOISE) - mu * mu;
    const float inv = __builtin_amdgcn_rsqf(var + 1e-5f);   // v_rsq_f32

    const float* cr = c + (size_t)row * (3 * NOISE);
#pragma unroll
    for (int e = 0; e < 2; ++e) {
        const int n   = t + e * 256;
        const float v = (e == 0) ? v0 : v1;
        const float xn = (v - mu) * inv;
        xm[(size_t)row * NOISE + n] = f2bf((1.0f + cr[n]) * xn + cr[NOISE + n]);
    }
}

// ---------------------------------------------------------------------------
// Host-side orchestration
// ---------------------------------------------------------------------------
extern "C" void kernel_launch(void* const* d_in, const int* in_sizes, int n_in,
                              void* d_out, int out_size, void* d_ws, size_t ws_size,
                              hipStream_t stream) {
    (void)in_sizes; (void)n_in; (void)out_size; (void)ws_size;

    const float* x_in  = (const float*)d_in[0];
    const float* timep = (const float*)d_in[1];
    const int*   cls   = (const int*)  d_in[2];
    const float* tbl   = (const float*)d_in[3];
    const float* emb_w = (const float*)d_in[4];
    const float* emb_b = (const float*)d_in[5];
    const float* cw1   = (const float*)d_in[6];
    const float* cb1   = (const float*)d_in[7];
    const float* cw2   = (const float*)d_in[8];
    const float* cb2   = (const float*)d_in[9];
    const float* mw1   = (const float*)d_in[10];
    const float* mb1   = (const float*)d_in[11];
    const float* mw2   = (const float*)d_in[12];
    const float* mb2   = (const float*)d_in[13];
    float* xout = (float*)d_out;

    // workspace layout
    char* ws = (char*)d_ws;
    unsigned short* cond_bf = (unsigned short*)ws;                       //  8 MB
    unsigned short* t1      = (unsigned short*)(ws + (size_t)BSZ * CONDD * 2);
    float*          cbuf    = (float*)(ws + (size_t)BSZ * CONDD * 2
                                          + (size_t)BSZ * LATENT * 2);   // 48 MB
    unsigned short* xm      = (unsigned short*)(ws + (size_t)BSZ * CONDD * 2
                                          + (size_t)BSZ * LATENT * 2
                                          + (size_t)BSZ * 3 * NOISE * 4);

    const dim3 blk(256);
    const int MT = BSZ / BM;   // 64 M tiles

    // cond = silu(emb_table[cls]) @ emb_w + emb_b + t_emb   -> bf16
    gemm_wmma<2><<<dim3(CONDD / BN, MT), blk, 0, stream>>>(
        tbl, emb_w, emb_b, cond_bf, BSZ, CONDD, LATENT,
        cls, timep, nullptr, nullptr);

    for (int i = 0; i < NBLK; ++i) {
        const float* xsrc = (i == 0) ? x_in : xout;

        // t1 = silu(cond @ cond_w1[i] + cond_b1[i])  -> bf16 [B, LATENT]
        gemm_wmma<0><<<dim3(LATENT / BN, MT), blk, 0, stream>>>(
            cond_bf, cw1 + (size_t)i * CONDD * LATENT, cb1 + (size_t)i * LATENT,
            t1, BSZ, LATENT, CONDD, nullptr, nullptr, nullptr, nullptr);

        // c = t1 @ cond_w2[i] + cond_b2[i]           -> f32 [B, 3*NOISE]
        gemm_wmma<1><<<dim3((3 * NOISE) / BN, MT), blk, 0, stream>>>(
            t1, cw2 + (size_t)i * LATENT * 3 * NOISE, cb2 + (size_t)i * 3 * NOISE,
            cbuf, BSZ, 3 * NOISE, LATENT, nullptr, nullptr, nullptr, nullptr);

        // xm = (1+scale1) * LN(x) + shift            -> bf16 [B, NOISE]
        ln_mod_kernel<<<BSZ, blk, 0, stream>>>(xsrc, cbuf, xm);

        // t1 = silu(xm @ mlp_w1[i] + mlp_b1[i])      -> bf16 [B, LATENT]
        gemm_wmma<0><<<dim3(LATENT / BN, MT), blk, 0, stream>>>(
            xm, mw1 + (size_t)i * NOISE * LATENT, mb1 + (size_t)i * LATENT,
            t1, BSZ, LATENT, NOISE, nullptr, nullptr, nullptr, nullptr);

        // x = res + (t1 @ mlp_w2[i] + mlp_b2[i]) * (1+scale2)/NBLK -> f32
        gemm_wmma<3><<<dim3(NOISE / BN, MT), blk, 0, stream>>>(
            t1, mw2 + (size_t)i * LATENT * NOISE, mb2 + (size_t)i * NOISE,
            xout, BSZ, NOISE, LATENT, nullptr, nullptr, cbuf, xsrc);
    }
}
